// GRUNet_36541581754561
// MI455X (gfx1250) — matrix operand order repaired and checked by reading, not compiled, not run on hardware
//
#include <hip/hip_runtime.h>
#include <hip/hip_bf16.h>
#include <stddef.h>

// GRU (2 layers) + FC head for MI455X (gfx1250, wave32, WMMA).
// v3: like v2 (register-resident weight B-fragments, double-buffered sX/sH,
// one barrier/step, x pipelined one step ahead) + fast gate activations via
// v_rcp_f32 instead of IEEE division on the sequential critical path.

#define B_TOT 512
#define T_LEN 512
#define H_DIM 128
#define NT_G  24      // 384 gate cols / 16
#define BT    16      // batch rows per block
#define KT_H  4       // 128 / 32 k-tiles for the recurrent GEMM

typedef __attribute__((ext_vector_type(16))) __bf16 v16bf;
typedef __attribute__((ext_vector_type(8)))  float  v8f;
typedef __attribute__((ext_vector_type(4)))  unsigned int v4u;

__device__ __forceinline__ unsigned short f2bf(float f) {
  union { float f; unsigned u; } c; c.f = f;
  unsigned u = c.u;
  return (unsigned short)((u + 0x7FFFu + ((u >> 16) & 1u)) >> 16);
}
__device__ __forceinline__ unsigned packbf2(float lo, float hi) {
  return (unsigned)f2bf(lo) | ((unsigned)f2bf(hi) << 16);
}
__device__ __forceinline__ v16bf load_frag(const unsigned* p) {
  union { v4u q[2]; v16bf v; } f;
  f.q[0] = *(const v4u*)p;
  f.q[1] = *(const v4u*)(p + 4);
  return f.v;
}
// Build one WMMA B-operand fragment (32x16, bf16) directly from global W[n][k]
// (row-major [384][K]). Lane layout: lane = (n&15) + 16*(kk>>4); vgpr v = K pair
// (2v,2v+1) of that half -> each lane packs 16 contiguous floats of row n.
__device__ __forceinline__ v16bf pack_frag_global(const float* __restrict__ W,
                                                  int K, int kt, int nt, int lane) {
  const int n = nt * 16 + (lane & 15);
  const int kbase = kt * 32 + ((lane >> 4) << 4);
  const float* p = W + n * K + kbase;
  union { unsigned d[8]; v16bf v; } f;
  #pragma unroll
  for (int i = 0; i < 8; ++i) f.d[i] = packbf2(p[2 * i], p[2 * i + 1]);
  return f.v;
}
// Gate activations on the sequential critical path: use v_exp_f32 + v_rcp_f32
// (hardware approx, ~1 ulp) instead of the IEEE div expansion.
__device__ __forceinline__ float fast_sigmoid(float x) {
  return __builtin_amdgcn_rcpf(1.0f + __expf(-x));
}
__device__ __forceinline__ float fast_tanh(float x) {
  float e = __expf(2.0f * x);
  return (e - 1.0f) * __builtin_amdgcn_rcpf(e + 1.0f);
}

// One full GRU layer over all T steps. x: [B, T, IN]. W*: [384, *] row-major.
// STORE_ALL: write h_t for every t (layout [B, T, H]); else only final h [B, H].
template<int IN, bool STORE_ALL>
__launch_bounds__(256, 1)
__global__ void gru_layer_kernel(const float* __restrict__ x,
                                 const float* __restrict__ Wih,
                                 const float* __restrict__ Whh,
                                 const float* __restrict__ bih,
                                 const float* __restrict__ bhh,
                                 float* __restrict__ hout)
{
  constexpr int KTI = IN / 32;
  constexpr bool WIH_REG = (IN == 64);              // layer0: W_ih fits in VGPRs too
  constexpr int WIH_LDS_DW = WIH_REG ? 0 : KTI * NT_G * 256;

  extern __shared__ unsigned smem[];
  unsigned* sWih = smem;                            // layer1 only: B-frags in LDS
  unsigned* sX   = smem + WIH_LDS_DW;               // 2 buffers * KTI*256 dw (A-frags)
  unsigned* sH   = sX + 2 * KTI * 256;              // 2 buffers * 1024 dw  (A-frags)

  const int tid  = threadIdx.x;
  const int w    = tid >> 5;
  const int lane = tid & 31;
  const int b0   = blockIdx.x * BT;

  // ---- recurrent weights -> registers (B-fragment layout, bf16) ----
  v16bf whh_r[KT_H], whh_z[KT_H], whh_n[KT_H];
  #pragma unroll
  for (int kt = 0; kt < KT_H; ++kt) {
    whh_r[kt] = pack_frag_global(Whh, H_DIM, kt, w,      lane);
    whh_z[kt] = pack_frag_global(Whh, H_DIM, kt, w + 8,  lane);
    whh_n[kt] = pack_frag_global(Whh, H_DIM, kt, w + 16, lane);
  }
  v16bf wih_r[KTI], wih_z[KTI], wih_n[KTI];
  if constexpr (WIH_REG) {
    #pragma unroll
    for (int kt = 0; kt < KTI; ++kt) {
      wih_r[kt] = pack_frag_global(Wih, IN, kt, w,      lane);
      wih_z[kt] = pack_frag_global(Wih, IN, kt, w + 8,  lane);
      wih_n[kt] = pack_frag_global(Wih, IN, kt, w + 16, lane);
    }
  } else {
    for (int d = tid; d < KTI * NT_G * 256; d += 256) {
      int v = d & 7, l = (d >> 3) & 31, frag = d >> 8;
      int kt = frag / NT_G, nt = frag % NT_G;
      int n = nt * 16 + (l & 15);
      int k = kt * 32 + ((l >> 4) << 4) + (v << 1);
      sWih[d] = packbf2(Wih[n * IN + k], Wih[n * IN + k + 1]);
    }
  }

  // Wave w owns gate N-tiles {w, w+8, w+16} -> r/z/n for h-cols [16w, 16w+16).
  // C layout: lane -> col (lane&15), vgpr v -> row v + 8*(lane>>4).
  const int jcol = (w << 4) + (lane & 15);
  const int mrow = (lane >> 4) << 3;
  const float bias_r  = bih[jcol]             + bhh[jcol];
  const float bias_z  = bih[H_DIM + jcol]     + bhh[H_DIM + jcol];
  const float bias_in = bih[2 * H_DIM + jcol];         // gi_n bias
  const float bias_hn = bhh[2 * H_DIM + jcol];         // gh_n bias (inside r*)

  v8f h_reg = {};   // persistent f32 hidden state: rows mrow..mrow+7, col jcol

  // h registers -> bf16 A-fragment layout in sH[buf]; our k-column is jcol.
  auto store_h_frag = [&](int buf) {
    unsigned short* s = (unsigned short*)(sH + buf * 1024);
    int kt = jcol >> 5, kk = jcol & 31, g = kk >> 3;
    int lhi = (kk >> 4) << 4;
    int va = (kk & 15) >> 1;
    int hi = kk & 1;
    #pragma unroll
    for (int v = 0; v < 8; ++v) {
      int m = v + mrow;
      s[((((kt << 5) + m + lhi) << 3) | va) * 2 + hi] = f2bf(h_reg[v]);
    }
  };
  // staged x values -> bf16 A-fragment layout in sX[buf]
  auto store_x_frag = [&](int buf, const float* xv) {
    unsigned short* s = (unsigned short*)(sX + buf * KTI * 256);
    #pragma unroll
    for (int i = 0; i < IN / 16; ++i) {
      int e = tid + (i << 8);
      int m = e / IN;
      int k = e % IN;
      int kt = k >> 5, kk = k & 31, g = kk >> 3;
      int la = m + ((kk >> 4) << 4);
      int va = (kk & 15) >> 1;
      s[((((kt << 5) + la) << 3) | va) * 2 + (kk & 1)] = f2bf(xv[i]);
    }
  };
  auto load_x = [&](int t, float* xv) {
    #pragma unroll
    for (int i = 0; i < IN / 16; ++i) {
      int e = tid + (i << 8);
      int m = e / IN;
      int k = e % IN;
      const float* p = &x[((size_t)(b0 + m) * T_LEN + t) * IN + k];
      xv[i] = STORE_ALL ? __builtin_nontemporal_load(p) : *p;  // layer0: x read-once
    }
  };

  // prologue: stage x_0 and h=0 into buffer 0
  {
    float x0[IN / 16];
    load_x(0, x0);
    store_x_frag(0, x0);
    store_h_frag(0);
  }
  __syncthreads();

  for (int t = 0; t < T_LEN; ++t) {
    const int rb = t & 1;
    const int wb = rb ^ 1;
    const unsigned* sXb = sX + rb * KTI * 256;
    const unsigned* sHb = sH + rb * 1024;

    // software pipeline: pull x_{t+1} into registers while WMMAs run
    float xnext[IN / 16];
    if (t + 1 < T_LEN) load_x(t + 1, xnext);
    if (t + 2 < T_LEN)
      __builtin_prefetch(&x[((size_t)(b0 + (tid >> 4)) * T_LEN + (t + 2)) * IN
                            + ((tid & 15) << 2)], 0, 1);

    v8f acc_r, acc_z, acc_in, acc_hn;
    #pragma unroll
    for (int v = 0; v < 8; ++v) {
      acc_r[v] = bias_r; acc_z[v] = bias_z; acc_in[v] = bias_in; acc_hn[v] = bias_hn;
    }

    // gi = x_t @ Wih^T  (r, z accumulate shared; n kept separate as gi_n)
    #pragma unroll
    for (int kt = 0; kt < KTI; ++kt) {
      v16bf a = load_frag(&sXb[((kt << 5) + lane) << 3]);
      v16bf br, bz, bn;
      if constexpr (WIH_REG) {
        br = wih_r[kt]; bz = wih_z[kt]; bn = wih_n[kt];
      } else {
        br = load_frag(&sWih[((((kt * NT_G) + w)      << 5) | lane) << 3]);
        bz = load_frag(&sWih[((((kt * NT_G) + w + 8)  << 5) | lane) << 3]);
        bn = load_frag(&sWih[((((kt * NT_G) + w + 16) << 5) | lane) << 3]);
      }
      acc_r  = __builtin_amdgcn_wmma_f32_16x16x32_bf16(false, a, false, br, (short)0, acc_r,  false, false);
      acc_z  = __builtin_amdgcn_wmma_f32_16x16x32_bf16(false, a, false, bz, (short)0, acc_z,  false, false);
      acc_in = __builtin_amdgcn_wmma_f32_16x16x32_bf16(false, a, false, bn, (short)0, acc_in, false, false);
    }
    // gh = h @ Whh^T with register-resident weights (gh_n separate: n = tanh(gi_n + r*gh_n))
    #pragma unroll
    for (int kt = 0; kt < KT_H; ++kt) {
      v16bf a = load_frag(&sHb[((kt << 5) + lane) << 3]);
      acc_r  = __builtin_amdgcn_wmma_f32_16x16x32_bf16(false, a, false, whh_r[kt], (short)0, acc_r,  false, false);
      acc_z  = __builtin_amdgcn_wmma_f32_16x16x32_bf16(false, a, false, whh_z[kt], (short)0, acc_z,  false, false);
      acc_hn = __builtin_amdgcn_wmma_f32_16x16x32_bf16(false, a, false, whh_n[kt], (short)0, acc_hn, false, false);
    }

    // gates + state update, fully in registers (v_exp/v_rcp, no IEEE div)
    #pragma unroll
    for (int v = 0; v < 8; ++v) {
      float r = fast_sigmoid(acc_r[v]);
      float z = fast_sigmoid(acc_z[v]);
      float n = fast_tanh(acc_in[v] + r * acc_hn[v]);
      h_reg[v] = (1.0f - z) * n + z * h_reg[v];
    }

    // commit next step's operands to the other buffer (no conflict with rb)
    if (t + 1 < T_LEN) {
      store_x_frag(wb, xnext);
      store_h_frag(wb);
    }
    __syncthreads();

    if (STORE_ALL) {  // h1 stays RT so layer-1 can consume it from the 192MB L2
      #pragma unroll
      for (int v = 0; v < 8; ++v) {
        int m = v + mrow;
        hout[((size_t)(b0 + m) * T_LEN + t) * H_DIM + jcol] = h_reg[v];
      }
    } else if (t == T_LEN - 1) {
      #pragma unroll
      for (int v = 0; v < 8; ++v) {
        int m = v + mrow;
        hout[(size_t)(b0 + m) * H_DIM + jcol] = h_reg[v];
      }
    }
  }
}

__global__ void fc_kernel(const float* __restrict__ h2, const float* __restrict__ Wfc,
                          const float* __restrict__ bfc, float* __restrict__ out)
{
  int b = blockIdx.x * blockDim.x + threadIdx.x;
  if (b >= B_TOT) return;
  float s = bfc[0];
  #pragma unroll 8
  for (int k = 0; k < H_DIM; ++k) s += h2[b * H_DIM + k] * Wfc[k];
  out[b] = s;
}

extern "C" void kernel_launch(void* const* d_in, const int* in_sizes, int n_in,
                              void* d_out, int out_size, void* d_ws, size_t ws_size,
                              hipStream_t stream) {
  (void)in_sizes; (void)n_in; (void)out_size; (void)ws_size;

  const float* x    = (const float*)d_in[0];
  const float* Wih0 = (const float*)d_in[1];
  const float* Whh0 = (const float*)d_in[2];
  const float* bih0 = (const float*)d_in[3];
  const float* bhh0 = (const float*)d_in[4];
  const float* Wih1 = (const float*)d_in[5];
  const float* Whh1 = (const float*)d_in[6];
  const float* bih1 = (const float*)d_in[7];
  const float* bhh1 = (const float*)d_in[8];
  const float* Wfc  = (const float*)d_in[9];
  const float* bfc  = (const float*)d_in[10];

  float* h1     = (float*)d_ws;                                   // [B, T, H] = 128 MB
  float* h2last = (float*)((char*)d_ws +
                  (size_t)B_TOT * T_LEN * H_DIM * sizeof(float)); // [B, H]

  const size_t lds0 = (size_t)(0              + 2 * 2 * 256 + 2 * 1024) * 4;  // 12 KB
  const size_t lds1 = (size_t)(4 * NT_G * 256 + 2 * 4 * 256 + 2 * 1024) * 4;  // 112 KB

  gru_layer_kernel<64,  true ><<<B_TOT / BT, 256, lds0, stream>>>(x,  Wih0, Whh0, bih0, bhh0, h1);
  gru_layer_kernel<128, false><<<B_TOT / BT, 256, lds1, stream>>>(h1, Wih1, Whh1, bih1, bhh1, h2last);
  fc_kernel<<<(B_TOT + 255) / 256, 256, 0, stream>>>(h2last, Wfc, bfc, (float*)d_out);
}
